// RGCN_50259707298098
// MI455X (gfx1250) — matrix operand order repaired and checked by reading, not compiled, not run on hardware
//
#include <hip/hip_runtime.h>
#include <hip/hip_bf16.h>

typedef __attribute__((ext_vector_type(16))) _Float16 v16h;
typedef __attribute__((ext_vector_type(8)))  float    v8f;
typedef __attribute__((ext_vector_type(4)))  float    v4f;

#define NN 8192          // nodes
#define RR 2             // relations
#define D0 128
#define D1 64
#define KTOT (RR * NN)   // flattened K = 16384

// ---------------------------------------------------------------------------
// Kernel 1: W0[r] = c0[r,0]*w0[0] + c0[r,1]*w0[1], f32 -> f16, written directly
// in WMMA B-fragment order:  B0[((kb*8 + j)*32 + lane)*16 + h]
//   kb = flat-K block (32 K per block, flatK = r*8192 + m), j = 16-col tile,
//   lane t: col = j*16 + (t&15), K_local = (t>>4)*16 + h
// ---------------------------------------------------------------------------
__global__ void rgcn_prep_w0(const float* __restrict__ w0,
                             const float* __restrict__ c0,
                             _Float16* __restrict__ B0) {
  int t    = blockIdx.x * blockDim.x + threadIdx.x;   // 131072 threads
  int lane = t & 31;
  int j    = (t >> 5) & 7;
  int kb   = (t >> 8) & 255;                          // K-block within r
  int r    = t >> 16;
  int col  = j * 16 + (lane & 15);
  int kHi  = (lane >> 4) * 16;
  float a0 = c0[r * 2 + 0];
  float a1 = c0[r * 2 + 1];
  _Float16* dst = B0 + ((size_t)t << 4);              // t encodes the full index
#pragma unroll
  for (int h = 0; h < 16; ++h) {
    int m = kb * 32 + kHi + h;                        // source node index
    float v = a0 * w0[(size_t)m * D0 + col] +
              a1 * w0[(size_t)NN * D0 + (size_t)m * D0 + col];
    dst[h] = (_Float16)v;
  }
}

// ---------------------------------------------------------------------------
// Main streaming GEMM:  C[8192 x DCOLS] = sum over flat K of adj_f16 * Bsw
//   - 256 threads = 8 waves: (strip = wave&1) x (kq = wave>>1, K-quarter)
//   - A: fp32 adj streamed non-temporally, converted to f16 in registers;
//        A-quad for step k+32 is prefetched one iteration ahead (HBM latency
//        hidden under current step's WMMAs)
//   - B: pre-swizzled f16 fragments (L2-resident). Issued BEFORE the next
//        A-quad so in-order LOADcnt retirement is not gated on HBM latency;
//        sched_group_barrier pins VMEM->VALU->WMMA order so wmma j runs at
//        loadcnt <= 2NT+2-2j instead of a full wait per wmma.
//   - cross-wave K reduction through LDS in a fixed order (deterministic)
// ---------------------------------------------------------------------------
template <int DCOLS>
__global__ __launch_bounds__(256)
void rgcn_gemm(const float* __restrict__ adj,
               const _Float16* __restrict__ Bsw,
               float* __restrict__ Cout) {
  constexpr int NT = DCOLS / 16;
  __shared__ float lds[8 * 16 * DCOLS];

  const int tid   = threadIdx.x;
  const int lane  = tid & 31;
  const int wave  = tid >> 5;
  const int strip = wave & 1;
  const int kq    = wave >> 1;                 // 0..3, each 4096 of flat K
  const int mBase = blockIdx.x * 32;
  const int row   = mBase + strip * 16 + (lane & 15);
  const int kSel  = lane >> 4;                 // lanes 16-31 hold K+16..K+31

  v8f acc[NT];
#pragma unroll
  for (int j = 0; j < NT; ++j)
#pragma unroll
    for (int v = 0; v < 8; ++v) acc[j][v] = 0.0f;

  const int k0   = kq * 4096;
  const int kEnd = k0 + 4096;
  const int rr   = kq >> 1;                    // each quarter is inside one r
  const float* base =
      adj + (((size_t)rr << 26) | ((size_t)row << 13)) + kSel * 16;

  // pipeline prologue: A-quad for the first step
  const float* ap = base + (k0 & (NN - 1));
  v4f f0 = __builtin_nontemporal_load((const v4f*)(ap));
  v4f f1 = __builtin_nontemporal_load((const v4f*)(ap + 4));
  v4f f2 = __builtin_nontemporal_load((const v4f*)(ap + 8));
  v4f f3 = __builtin_nontemporal_load((const v4f*)(ap + 12));

#pragma unroll 2
  for (int kk = k0; kk < kEnd; kk += 32) {
    // current-step B fragments, issued first (fast L2 returns retire first)
    const int kb = kk >> 5;
    const _Float16* bp = Bsw + ((((size_t)kb * NT) * 32 + lane) << 4);
    v16h b[NT];
#pragma unroll
    for (int j = 0; j < NT; ++j)
      b[j] = *(const v16h*)(bp + (((size_t)j * 32) << 4));

    // prefetch next A-quad (last step wraps to k0: harmless in-range re-read)
    const int kn = (kk + 32 < kEnd) ? (kk + 32) : k0;
    const float* apn = base + (kn & (NN - 1));
    v4f g0 = __builtin_nontemporal_load((const v4f*)(apn));
    v4f g1 = __builtin_nontemporal_load((const v4f*)(apn + 4));
    v4f g2 = __builtin_nontemporal_load((const v4f*)(apn + 8));
    v4f g3 = __builtin_nontemporal_load((const v4f*)(apn + 12));

    // convert the A-quad prefetched last iteration
    v16h a;
#pragma unroll
    for (int e = 0; e < 4; ++e) {
      a[e]      = (_Float16)f0[e];
      a[4 + e]  = (_Float16)f1[e];
      a[8 + e]  = (_Float16)f2[e];
      a[12 + e] = (_Float16)f3[e];
    }
#pragma unroll
    for (int j = 0; j < NT; ++j)
      acc[j] = __builtin_amdgcn_wmma_f32_16x16x32_f16(
          false, a, false, b[j], (short)0, acc[j], false, false);

    f0 = g0; f1 = g1; f2 = g2; f3 = g3;

    // pin per-step schedule: all VMEM reads, then cvts, then WMMAs
    __builtin_amdgcn_sched_group_barrier(0x0020, 2 * NT + 4, 0); // VMEM read
    __builtin_amdgcn_sched_group_barrier(0x0002, 8, 0);          // VALU (cvt)
    __builtin_amdgcn_sched_group_barrier(0x0008, NT, 0);         // WMMA
  }

  // C fragment layout: lane c<16 -> col=c, rows v; lane c+16 -> col=c-16, rows 8+v
  float* myLds = lds + wave * (16 * DCOLS);
  const int mOff = kSel * 8;
  const int cl   = lane & 15;
#pragma unroll
  for (int j = 0; j < NT; ++j)
#pragma unroll
    for (int v = 0; v < 8; ++v)
      myLds[(mOff + v) * DCOLS + (j * 16 + cl)] = acc[j][v];
  __syncthreads();

  for (int o = tid; o < 32 * DCOLS; o += 256) {
    int ml  = o / DCOLS;
    int col = o - ml * DCOLS;
    int st  = ml >> 4;
    int mr  = ml & 15;
    float s = 0.0f;
#pragma unroll
    for (int q = 0; q < 4; ++q)
      s += lds[(q * 2 + st) * (16 * DCOLS) + mr * DCOLS + col];
    Cout[(size_t)(mBase + ml) * DCOLS + col] = s;
  }
}

// ---------------------------------------------------------------------------
// Kernel 3: hW1[r] = relu(out0) @ (c1[r,0]*w1[0] + c1[r,1]*w1[1])   [8192 x 64]
// grid = (128, 2); LDS-staged h tile (32KB) and combined W1 (32KB).
// ---------------------------------------------------------------------------
__global__ __launch_bounds__(256)
void rgcn_hw1(const float* __restrict__ out0,
              const float* __restrict__ w1,
              const float* __restrict__ c1,
              float* __restrict__ hW1) {
  __shared__ float hs[64 * D0];
  __shared__ float wc[D0 * D1];
  const int tid   = threadIdx.x;
  const int r     = blockIdx.y;
  const int mBase = blockIdx.x * 64;
  for (int i = tid; i < 64 * D0; i += 256) {
    float v = out0[(size_t)mBase * D0 + i];
    hs[i] = v > 0.0f ? v : 0.0f;
  }
  const float ca = c1[r * 2 + 0];
  const float cb = c1[r * 2 + 1];
  for (int i = tid; i < D0 * D1; i += 256)
    wc[i] = ca * w1[i] + cb * w1[D0 * D1 + i];
  __syncthreads();

  const int lane = tid & 31;
  const int wave = tid >> 5;
  for (int mi = 0; mi < 8; ++mi) {
    const int mloc = wave * 8 + mi;            // uniform per wave -> LDS broadcast
#pragma unroll
    for (int pp = 0; pp < 2; ++pp) {
      const int p = pp * 32 + lane;            // conflict-free on wc
      float s = 0.0f;
#pragma unroll 8
      for (int d = 0; d < D0; ++d)
        s += hs[mloc * D0 + d] * wc[d * D1 + p];
      hW1[((size_t)r * NN + mBase + mloc) * D1 + p] = s;
    }
  }
}

// ---------------------------------------------------------------------------
// Kernel 4: swizzle hW1 (f32 row-major) into f16 WMMA B-fragment layout (D1=64)
// ---------------------------------------------------------------------------
__global__ void rgcn_swz1(const float* __restrict__ hW1, _Float16* __restrict__ B1) {
  int t    = blockIdx.x * blockDim.x + threadIdx.x;   // 65536 threads
  int lane = t & 31;
  int j    = (t >> 5) & 3;
  int kb   = (t >> 7) & 255;
  int r    = t >> 15;
  int p    = j * 16 + (lane & 15);
  int kHi  = (lane >> 4) * 16;
  const float* src = hW1 + (size_t)r * NN * D1;
  _Float16* dst = B1 + ((size_t)t << 4);
#pragma unroll
  for (int h = 0; h < 16; ++h) {
    int m = kb * 32 + kHi + h;
    dst[h] = (_Float16)src[(size_t)m * D1 + p];
  }
}

// ---------------------------------------------------------------------------
// final_rep reduction: fixed-order two-stage column sum (no float atomics)
// ---------------------------------------------------------------------------
__global__ void rgcn_colsum_part(const float* __restrict__ out0,
                                 const float* __restrict__ out1,
                                 float* __restrict__ parts) {
  const int c = threadIdx.x;                   // 0..191
  const int g = blockIdx.x;                    // 0..63, 128 rows each
  float s = 0.0f;
  if (c < D0) {
    for (int i = 0; i < 128; ++i) s += out0[(size_t)(g * 128 + i) * D0 + c];
  } else {
    const int p = c - D0;
    for (int i = 0; i < 128; ++i) s += out1[(size_t)(g * 128 + i) * D1 + p];
  }
  parts[g * 192 + c] = s;
}

__global__ void rgcn_colsum_final(const float* __restrict__ parts,
                                  float* __restrict__ dst) {
  const int c = threadIdx.x;                   // 0..191
  float s = 0.0f;
  for (int g = 0; g < 64; ++g) s += parts[g * 192 + c];
  dst[c] = s;
}

// ---------------------------------------------------------------------------
extern "C" void kernel_launch(void* const* d_in, const int* in_sizes, int n_in,
                              void* d_out, int out_size, void* d_ws, size_t ws_size,
                              hipStream_t stream) {
  const float* adj = (const float*)d_in[0];   // [2, 8192, 8192]
  const float* w0  = (const float*)d_in[1];   // [2, 8192, 128]
  const float* c0  = (const float*)d_in[2];   // [2, 2]
  const float* w1  = (const float*)d_in[3];   // [2, 128, 64]
  const float* c1  = (const float*)d_in[4];   // [2, 2]
  float* out = (float*)d_out;                 // out1 [8192,64] ++ final_rep [192]

  char* ws = (char*)d_ws;
  _Float16* B0   = (_Float16*)(ws);                      // 4 MB swizzled W0 f16
  float*    out0 = (float*)(ws + ((size_t)4 << 20));     // 4 MB [8192,128]
  float*    hW1  = (float*)(ws + ((size_t)8 << 20));     // 4 MB [2,8192,64]
  _Float16* B1   = (_Float16*)(ws + ((size_t)12 << 20)); // 2 MB swizzled hW1 f16
  float*    prt  = (float*)(ws + ((size_t)14 << 20));    // 48 KB partial colsums

  // 1) W0 basis combine + f16 swizzle
  rgcn_prep_w0<<<512, 256, 0, stream>>>(w0, c0, B0);
  // 2) out0 = sum_r adj[r] @ W0[r]      (pass 1 over adj, 512 MB NT stream)
  rgcn_gemm<D0><<<256, 256, 0, stream>>>(adj, B0, out0);
  // 3) hW1[r] = relu(out0) @ W1[r]
  rgcn_hw1<<<dim3(128, 2), 256, 0, stream>>>(out0, w1, c1, hW1);
  // 4) swizzle hW1 to f16 fragments
  rgcn_swz1<<<256, 256, 0, stream>>>(hW1, B1);
  // 5) out1 = sum_r adj[r] @ hW1[r]     (pass 2 over adj)
  rgcn_gemm<D1><<<256, 256, 0, stream>>>(adj, B1, out);
  // 6/7) final_rep = column sums of [out0 | out1]
  rgcn_colsum_part<<<64, 192, 0, stream>>>(out0, out, prt);
  rgcn_colsum_final<<<1, 192, 0, stream>>>(prt, out + (size_t)NN * D1);
}
